// DGCNNClassifier_4011499454874
// MI455X (gfx1250) — matrix operand order repaired and checked by reading, compiled, not run on hardware
//
#include <hip/hip_runtime.h>
#include <hip/hip_bf16.h>

#define EPSV   1e-5f
#define SLOPE  0.2f
#define BATCH  8
#define NPTS   1024
#define MROWS  (BATCH * NPTS)   // 8192
#define KNN    20
#define NEGINF (-3.0e38f)

typedef _Float16 f16;
typedef __attribute__((ext_vector_type(8)))  f16   v8h;
typedef __attribute__((ext_vector_type(16))) f16   v16h;
typedef __attribute__((ext_vector_type(8)))  float v8f;

// ---------------------------------------------------------------------------
// WMMA fragment loaders (CDNA5 wave32 layouts, ISA 7.12.2)
// A: 16x32 f16, row-major source [*, lda]; lane L -> M=L%16,
//    per lane two contiguous 8-half (16B) loads at kk+8*(L>>4) and +16.
// B: 32x16 f16 from K-major buffer BT[N][Kp]; lane L -> N=L%16,
//    per lane two contiguous 8-half (16B) loads at kk+8*(L>>4) and +16.
// ---------------------------------------------------------------------------
static __device__ __forceinline__ v16h frag_a(const f16* base, int lda, int row16, int kk, int lane) {
    const f16* p = base + (size_t)(row16 + (lane & 15)) * lda + kk + ((lane >> 4) << 3);
    v8h lo = *(const v8h*)p;
    v8h hi = *(const v8h*)(p + 16);
    return __builtin_shufflevector(lo, hi, 0,1,2,3,4,5,6,7,8,9,10,11,12,13,14,15);
}

static __device__ __forceinline__ v16h frag_b(const f16* base, int ldb, int col16, int kk, int lane) {
    const f16* p = base + (size_t)(col16 + (lane & 15)) * ldb + kk + ((lane >> 4) << 3);
    v8h lo = *(const v8h*)p;
    v8h hi = *(const v8h*)(p + 16);
    return __builtin_shufflevector(lo, hi, 0,1,2,3,4,5,6,7,8,9,10,11,12,13,14,15);
}

// ---------------------------------------------------------------------------
// Prep: points f32 [8192,3] -> A0 f16 [8192,32] (zero-padded K), xx = ||x||^2
// ---------------------------------------------------------------------------
__global__ void prep_points_kernel(const float* __restrict__ pts, f16* __restrict__ a0,
                                   float* __restrict__ xx) {
    int id = blockIdx.x * blockDim.x + threadIdx.x;
    if (id >= MROWS * 32) return;
    int p = id >> 5, kk = id & 31;
    float v = (kk < 3) ? pts[p * 3 + kk] : 0.0f;
    a0[p * 32 + kk] = (f16)v;
    if (kk == 0) {
        float x0 = pts[p*3], x1 = pts[p*3+1], x2 = pts[p*3+2];
        xx[p] = x0*x0 + x1*x1 + x2*x2;
    }
}

// W [O,2C] f32 -> WaT [O,Kp] f16 (cols 0..C-1) and WdT [O,Kp] = Wb - Wa
__global__ void prep_w_pair_kernel(const float* __restrict__ W, f16* __restrict__ waT,
                                   f16* __restrict__ wdT, int O, int C, int Kp) {
    int id = blockIdx.x * blockDim.x + threadIdx.x;
    if (id >= O * Kp) return;
    int o = id / Kp, kq = id - o * Kp;
    float wa = 0.0f, wd = 0.0f;
    if (kq < C) {
        wa = W[o * 2 * C + kq];
        wd = W[o * 2 * C + C + kq] - wa;
    }
    waT[id] = (f16)wa;
    wdT[id] = (f16)wd;
}

__global__ void prep_w_single_kernel(const float* __restrict__ W, f16* __restrict__ wT, int n) {
    int id = blockIdx.x * blockDim.x + threadIdx.x;
    if (id < n) wT[id] = (f16)W[id];
}

// ---------------------------------------------------------------------------
// Gram/score: scores[b][n][m] = 2 * x_n . x_m - ||x_m||^2   (rank-equivalent
// to -||x_n - x_m||^2 for fixed n). One wave per 16x64 strip: the A (row-n)
// fragment is loaded once per K-step and reused across 4 m-tiles.
// ---------------------------------------------------------------------------
__global__ void wmma_gram_kernel(const f16* __restrict__ X, int lda, int Kp,
                                 const float* __restrict__ xx, float* __restrict__ scores) {
    int wave = (blockIdx.x * blockDim.x + threadIdx.x) >> 5;
    int lane = threadIdx.x & 31;
    if (wave >= BATCH * 64 * 16) return;       // 64 n-tiles x 16 m-quads per batch
    int b  = wave >> 10;
    int rm = wave & 1023;
    int nt = rm >> 4, m4 = rm & 15;
    int rowA = b * NPTS + nt * 16;
    int rowB = b * NPTS + m4 * 64;

    v8f acc[4];
    for (int j = 0; j < 4; ++j)
        for (int i = 0; i < 8; ++i) acc[j][i] = 0.0f;

    for (int kk = 0; kk < Kp; kk += 32) {
        v16h a = frag_a(X, lda, rowA, kk, lane);
        #pragma unroll
        for (int j = 0; j < 4; ++j) {
            v16h bf = frag_b(X, lda, rowB + j * 16, kk, lane);
            acc[j] = __builtin_amdgcn_wmma_f32_16x16x32_f16(false, a, false, bf, (short)0,
                                                            acc[j], false, false);
        }
    }
    int nb = nt * 16 + ((lane >> 4) << 3);
    #pragma unroll
    for (int j = 0; j < 4; ++j) {
        int m = m4 * 64 + j * 16 + (lane & 15);
        float xm = xx[b * NPTS + m];
        float* srow = scores + ((size_t)b << 20) + (size_t)m;
        for (int r = 0; r < 8; ++r)
            srow[(size_t)(nb + r) * NPTS] = 2.0f * acc[j][r] - xm;
    }
}

// ---------------------------------------------------------------------------
// Top-k (k=20): wave per row, iterative wave-wide argmax with shfl_xor,
// lower-index tie break (matches jax.lax.top_k ordering semantics).
// ---------------------------------------------------------------------------
__global__ void topk_kernel(const float* __restrict__ scores, int* __restrict__ idx) {
    int row  = (blockIdx.x * blockDim.x + threadIdx.x) >> 5;
    int lane = threadIdx.x & 31;
    if (row >= MROWS) return;
    const float* srow = scores + (size_t)row * NPTS;
    float vals[32];
    #pragma unroll
    for (int i = 0; i < 32; ++i) vals[i] = srow[lane + 32 * i];

    for (int t = 0; t < KNN; ++t) {
        float best = NEGINF; int bi = 0x7fffffff;
        #pragma unroll
        for (int i = 0; i < 32; ++i) {
            float v = vals[i];
            int   m = lane + 32 * i;
            if (v > best || (v == best && m < bi)) { best = v; bi = m; }
        }
        #pragma unroll
        for (int off = 16; off > 0; off >>= 1) {
            float ov = __shfl_xor(best, off, 32);
            int   oi = __shfl_xor(bi,   off, 32);
            if (ov > best || (ov == best && oi < bi)) { best = ov; bi = oi; }
        }
        if (lane == 0) idx[row * KNN + t] = bi;
        if ((bi & 31) == lane) vals[bi >> 5] = NEGINF;
    }
}

// ---------------------------------------------------------------------------
// WMMA GEMM: out[M,O] = A[M,Kp](f16, stride lda) * BT[O,Kp]^T, f32 accum.
// Register-blocked: one wave computes a 16x64 strip (1 m-tile x 4 n-tiles),
// A fragment loaded once per K-step, reused for 4 WMMAs.
// Optional fused BN + leaky-ReLU epilogue (per output column).
// ---------------------------------------------------------------------------
__global__ void wmma_gemm_kernel(const f16* __restrict__ A, int lda,
                                 const f16* __restrict__ BT, int ldb,
                                 float* __restrict__ out, int ldo,
                                 int nquads, int total_waves, int Kp,
                                 const float* __restrict__ g, const float* __restrict__ bb,
                                 const float* __restrict__ mm, const float* __restrict__ vv) {
    int wave = (blockIdx.x * blockDim.x + threadIdx.x) >> 5;
    int lane = threadIdx.x & 31;
    if (wave >= total_waves) return;
    int mt = wave / nquads, n4 = wave - mt * nquads;

    v8f acc[4];
    for (int j = 0; j < 4; ++j)
        for (int i = 0; i < 8; ++i) acc[j][i] = 0.0f;

    for (int kk = 0; kk < Kp; kk += 32) {
        if (kk + 32 < Kp)
            __builtin_prefetch(A + (size_t)(mt * 16 + (lane & 15)) * lda + kk + 32, 0, 1);
        v16h a = frag_a(A, lda, mt * 16, kk, lane);
        #pragma unroll
        for (int j = 0; j < 4; ++j) {
            v16h bf = frag_b(BT, ldb, n4 * 64 + j * 16, kk, lane);
            acc[j] = __builtin_amdgcn_wmma_f32_16x16x32_f16(false, a, false, bf, (short)0,
                                                            acc[j], false, false);
        }
    }
    int rb = mt * 16 + ((lane >> 4) << 3);
    #pragma unroll
    for (int j = 0; j < 4; ++j) {
        int o = n4 * 64 + j * 16 + (lane & 15);
        if (g) {
            float s  = g[o] * rsqrtf(vv[o] + EPSV);
            float bi = bb[o] - mm[o] * s;
            for (int r = 0; r < 8; ++r) {
                float h = acc[j][r] * s + bi;
                h = (h > 0.0f) ? h : SLOPE * h;
                out[(size_t)(rb + r) * ldo + o] = h;
            }
        } else {
            for (int r = 0; r < 8; ++r)
                out[(size_t)(rb + r) * ldo + o] = acc[j][r];
        }
    }
}

// ---------------------------------------------------------------------------
// Gather-max epilogue of edge conv: y[p,o] = max_kk lrelu(bn(u[j,o] + t[p,o]))
// Writes f16 into the concat column slice (row stride 512) and ||y||^2 -> xx.
// Block = one point, threads = O.
// ---------------------------------------------------------------------------
__global__ void gathermax_kernel(const float* __restrict__ u, const float* __restrict__ t,
                                 const int* __restrict__ idx,
                                 const float* __restrict__ g, const float* __restrict__ bb,
                                 const float* __restrict__ mm, const float* __restrict__ vv,
                                 f16* __restrict__ ycat, int ldcat,
                                 float* __restrict__ xxout, int O) {
    __shared__ float red[256];
    int p = blockIdx.x;
    int o = threadIdx.x;
    float s  = g[o] * rsqrtf(vv[o] + EPSV);
    float bi = bb[o] - mm[o] * s;
    float tv = t[(size_t)p * O + o];
    int base = (p >> 10) << 10;  // batch base row
    const int* ip = idx + p * KNN;
    float best = NEGINF;
    #pragma unroll 4
    for (int kk = 0; kk < KNN; ++kk) {
        int j = base + ip[kk];
        float h = u[(size_t)j * O + o] + tv;
        h = h * s + bi;
        h = (h > 0.0f) ? h : SLOPE * h;
        best = fmaxf(best, h);
    }
    ycat[(size_t)p * ldcat + o] = (f16)best;

    red[o] = best * best;
    __syncthreads();
    for (int st = blockDim.x >> 1; st > 0; st >>= 1) {
        if (o < st) red[o] += red[o + st];
        __syncthreads();
    }
    if (o == 0) xxout[p] = red[0];
}

// ---------------------------------------------------------------------------
// Global max + mean pooling over N: xg [8192,1024] -> feat [8, 2048]
// grid (4 col-chunks, 8 batches), 256 threads over columns (coalesced rows).
// ---------------------------------------------------------------------------
__global__ void pool_kernel(const float* __restrict__ xg, float* __restrict__ feat) {
    int o = blockIdx.x * 256 + threadIdx.x;
    int b = blockIdx.y;
    float mx = NEGINF, sm = 0.0f;
    for (int n = 0; n < NPTS; ++n) {
        float v = xg[(size_t)(b * NPTS + n) * 1024 + o];
        mx = fmaxf(mx, v);
        sm += v;
    }
    feat[b * 2048 + o]        = mx;
    feat[b * 2048 + 1024 + o] = sm * (1.0f / NPTS);
}

// ---------------------------------------------------------------------------
// Small FC: out[R,O] = in[R,K] @ W[O,K]^T, then BN+lrelu (g!=null) or +bias.
// R=8 so scalar f32 dot products are fine (not WMMA-shaped).
// ---------------------------------------------------------------------------
__global__ void fc_kernel(const float* __restrict__ in, int R, int K,
                          const float* __restrict__ W, int O,
                          const float* __restrict__ g, const float* __restrict__ bb,
                          const float* __restrict__ mm, const float* __restrict__ vv,
                          const float* __restrict__ bias, float* __restrict__ out) {
    int id = blockIdx.x * blockDim.x + threadIdx.x;
    if (id >= R * O) return;
    int r = id / O, o = id - r * O;
    const float* ip = in + (size_t)r * K;
    const float* wp = W + (size_t)o * K;
    float acc = 0.0f;
    for (int k = 0; k < K; ++k) acc += ip[k] * wp[k];
    if (g) {
        float s = g[o] * rsqrtf(vv[o] + EPSV);
        acc = acc * s + (bb[o] - mm[o] * s);
        acc = (acc > 0.0f) ? acc : SLOPE * acc;
    } else {
        acc += bias[o];
    }
    out[(size_t)r * O + o] = acc;
}

// ---------------------------------------------------------------------------
extern "C" void kernel_launch(void* const* d_in, const int* in_sizes, int n_in,
                              void* d_out, int out_size, void* d_ws, size_t ws_size,
                              hipStream_t stream) {
    (void)in_sizes; (void)n_in; (void)out_size; (void)ws_size;

    const float* pts = (const float*)d_in[0];
    // params flattened in dict insertion order: ec1..ec4 {W,g,b,m,v}, fuse, fc1, fc2 {W,g,b,m,v}, fc3 {W,bias}
    const float* ecW[4], *ecG[4], *ecB[4], *ecM[4], *ecV[4];
    for (int i = 0; i < 4; ++i) {
        ecW[i] = (const float*)d_in[1 + 5*i + 0];
        ecG[i] = (const float*)d_in[1 + 5*i + 1];
        ecB[i] = (const float*)d_in[1 + 5*i + 2];
        ecM[i] = (const float*)d_in[1 + 5*i + 3];
        ecV[i] = (const float*)d_in[1 + 5*i + 4];
    }
    const float* fuW = (const float*)d_in[21], *fuG = (const float*)d_in[22],
               * fuB = (const float*)d_in[23], *fuM = (const float*)d_in[24],
               * fuV = (const float*)d_in[25];
    const float* f1W = (const float*)d_in[26], *f1G = (const float*)d_in[27],
               * f1B = (const float*)d_in[28], *f1M = (const float*)d_in[29],
               * f1V = (const float*)d_in[30];
    const float* f2W = (const float*)d_in[31], *f2G = (const float*)d_in[32],
               * f2B = (const float*)d_in[33], *f2M = (const float*)d_in[34],
               * f2V = (const float*)d_in[35];
    const float* f3W = (const float*)d_in[36], *f3Bias = (const float*)d_in[37];

    // workspace carve-up
    char* base = (char*)d_ws;
    size_t off = 0;
    auto carve = [&](size_t bytes) { char* p = base + off; off = (off + bytes + 255) & ~(size_t)255; return p; };
    f16*   A0     = (f16*)  carve((size_t)MROWS * 32 * 2);        // points, Kp=32
    f16*   concat = (f16*)  carve((size_t)MROWS * 512 * 2);       // [x1|x2|x3|x4] f16
    float* xx     = (float*)carve((size_t)MROWS * 4);
    float* scores = (float*)carve((size_t)BATCH * NPTS * NPTS * 4); // reused as xg
    int*   nidx   = (int*)  carve((size_t)MROWS * KNN * 4);
    float* ubuf   = (float*)carve((size_t)MROWS * 256 * 4);
    float* tbuf   = (float*)carve((size_t)MROWS * 256 * 4);
    f16*   wa[4], *wd[4];
    const int ecC[4]  = {3, 64, 64, 128};   // input channels
    const int ecKp[4] = {32, 64, 64, 128};  // padded K
    const int ecO[4]  = {64, 64, 128, 256}; // output channels
    for (int i = 0; i < 4; ++i) {
        wa[i] = (f16*)carve((size_t)ecO[i] * ecKp[i] * 2);
        wd[i] = (f16*)carve((size_t)ecO[i] * ecKp[i] * 2);
    }
    f16*   wfuse = (f16*)  carve((size_t)1024 * 512 * 2);
    float* feat  = (float*)carve((size_t)BATCH * 2048 * 4);
    float* f1o   = (float*)carve((size_t)BATCH * 512 * 4);
    float* f2o   = (float*)carve((size_t)BATCH * 256 * 4);

    // --- prep ---
    prep_points_kernel<<<(MROWS * 32) / 256, 256, 0, stream>>>(pts, A0, xx);
    for (int i = 0; i < 4; ++i) {
        int n = ecO[i] * ecKp[i];
        prep_w_pair_kernel<<<(n + 255) / 256, 256, 0, stream>>>(ecW[i], wa[i], wd[i],
                                                                ecO[i], ecC[i], ecKp[i]);
    }
    prep_w_single_kernel<<<(1024 * 512) / 256, 256, 0, stream>>>(fuW, wfuse, 1024 * 512);

    // --- 4 edge-conv blocks ---
    const int colOff[4] = {0, 0, 64, 128};     // input column slice in concat (block 0 uses A0)
    const int outOff[4] = {0, 64, 128, 256};   // output column slice in concat
    for (int i = 0; i < 4; ++i) {
        const f16* X  = (i == 0) ? A0 : (concat + colOff[i]);
        int        ld = (i == 0) ? 32 : 512;
        int Kp = ecKp[i], O = ecO[i];

        // pairwise scores via WMMA gram (16x64 strips per wave)
        wmma_gram_kernel<<<(BATCH * 64 * 16) / 4, 128, 0, stream>>>(X, ld, Kp, xx, scores);
        // top-k neighbor indices
        topk_kernel<<<MROWS / 4, 128, 0, stream>>>(scores, nidx);
        // u = X * Wa^T, t = X * (Wb-Wa)^T  (16x64 strips per wave)
        int nq = O / 64, waves = (MROWS / 16) * nq;
        wmma_gemm_kernel<<<waves / 4, 128, 0, stream>>>(X, ld, wa[i], Kp, ubuf, O,
                                                        nq, waves, Kp,
                                                        nullptr, nullptr, nullptr, nullptr);
        wmma_gemm_kernel<<<waves / 4, 128, 0, stream>>>(X, ld, wd[i], Kp, tbuf, O,
                                                        nq, waves, Kp,
                                                        nullptr, nullptr, nullptr, nullptr);
        // gather + BN + lrelu + max over neighbors -> concat slice + xx for next block
        gathermax_kernel<<<MROWS, O, 0, stream>>>(ubuf, tbuf, nidx,
                                                  ecG[i], ecB[i], ecM[i], ecV[i],
                                                  concat + outOff[i], 512, xx, O);
    }

    // --- fuse GEMM [8192,512] x [512,1024] with fused BN+lrelu; xg reuses scores ---
    float* xg = scores;
    {
        int nq = 1024 / 64, waves = (MROWS / 16) * nq;
        wmma_gemm_kernel<<<waves / 4, 128, 0, stream>>>(concat, 512, wfuse, 512, xg, 1024,
                                                        nq, waves, 512,
                                                        fuG, fuB, fuM, fuV);
    }

    // --- pooling -> feat [8, 2048] ---
    {
        dim3 grid(4, BATCH);
        pool_kernel<<<grid, 256, 0, stream>>>(xg, feat);
    }

    // --- FCs ---
    fc_kernel<<<(BATCH * 512 + 127) / 128, 128, 0, stream>>>(feat, BATCH, 2048, f1W, 512,
                                                             f1G, f1B, f1M, f1V, nullptr, f1o);
    fc_kernel<<<(BATCH * 256 + 127) / 128, 128, 0, stream>>>(f1o, BATCH, 512, f2W, 256,
                                                             f2G, f2B, f2M, f2V, nullptr, f2o);
    fc_kernel<<<(BATCH * 40 + 127) / 128, 128, 0, stream>>>(f2o, BATCH, 256, f3W, 40,
                                                            nullptr, nullptr, nullptr, nullptr,
                                                            f3Bias, (float*)d_out);
}
